// TemporalAttention_54606214202018
// MI455X (gfx1250) — matrix operand order repaired and checked by reading, not compiled, not run on hardware
//
#include <hip/hip_runtime.h>
#include <hip/hip_bf16.h>

// ---------------------------------------------------------------------------
// CDNA5 / gfx1250 implementation. wave32 only. All GEMMs via
// v_wmma_f32_16x16x32_bf16 (bf16 A/B, f32 accumulate).
// Softmaxes use a fixed max bound (scores <= SCALE since q,k unit-norm;
// sim values <= 1) -> exact softmax math, no online-max machinery.
// ---------------------------------------------------------------------------

typedef __attribute__((ext_vector_type(16))) __bf16 v16bf;
typedef __attribute__((ext_vector_type(8)))  __bf16 v8bf;
typedef __attribute__((ext_vector_type(8)))  float  v8f;

#define NTOK   4096
#define CDIM   256
#define HEADS  4
#define DHEAD  64
#define TCDIM  768
#define SCALE_F 25.0f

__device__ __forceinline__ v8f wmma_bf16(v16bf a, v16bf b, v8f c) {
  return __builtin_amdgcn_wmma_f32_16x16x32_bf16(
      /*neg_a=*/false, a, /*neg_b=*/false, b,
      /*c_mod=*/(short)0, c, /*reuse_a=*/false, /*reuse_b=*/false);
}

// A-matrix 16x32 bf16 fragment from a row-major row pointer.
// Contiguous layout: elem e -> K = kbase + 16*half + e
__device__ __forceinline__ v16bf load_afrag(const __bf16* row, int kbase, int half) {
  const v8bf lo = *(const v8bf*)(row + kbase + 16 * half);
  const v8bf hi = *(const v8bf*)(row + kbase + 16 * half + 8);
  v16bf a;
#pragma unroll
  for (int e = 0; e < 8; ++e) { a[e] = lo[e]; a[8 + e] = hi[e]; }
  return a;
}

__device__ __forceinline__ float rsum16(float v) {
  v += __shfl_xor(v, 1, 32);
  v += __shfl_xor(v, 2, 32);
  v += __shfl_xor(v, 4, 32);
  v += __shfl_xor(v, 8, 32);
  return v;
}
__device__ __forceinline__ float rsum32(float v) {
  v += __shfl_xor(v, 16, 32);
  return rsum16(v);
}

// ---------------------------------------------------------------------------
// Prep kernels: one-time f32 -> bf16 conversion; W transposed to [768][256]
// so B-fragments are contiguous per-lane 32B chunks.
// ---------------------------------------------------------------------------
__global__ __launch_bounds__(256) void k_cvt_x(
    const float* __restrict__ x_c, const float* __restrict__ x_r,
    __bf16* __restrict__ xb_c, __bf16* __restrict__ xb_r) {
  const int gid = blockIdx.x * 256 + threadIdx.x;   // 2 * N * C threads
  const int i = gid & (NTOK * CDIM - 1);
  if (gid < NTOK * CDIM) xb_c[i] = (__bf16)x_c[i];
  else                   xb_r[i] = (__bf16)x_r[i];
}

__global__ __launch_bounds__(256) void k_cvt_wt(
    const float* __restrict__ W_c, const float* __restrict__ W_r,
    __bf16* __restrict__ Wt_c, __bf16* __restrict__ Wt_r) {
  const int gid = blockIdx.x * 256 + threadIdx.x;   // 2 * 768 * 256 threads
  const int i = gid & (TCDIM * CDIM - 1);
  const int m = i >> 8, k = i & 255;                // Wt[m][k] = W[k][m]
  if (gid < TCDIM * CDIM) Wt_c[i] = (__bf16)W_c[(size_t)k * TCDIM + m];
  else                    Wt_r[i] = (__bf16)W_r[(size_t)k * TCDIM + m];
}

// ---------------------------------------------------------------------------
// Kernel 1: QKV projection  qkv[n][m] = sum_c x[n][c] * W[c][m]
// grid (N/16, 768/64, 2), block 32: one wave -> 16x64 strip (A reused 4x)
// ---------------------------------------------------------------------------
__global__ __launch_bounds__(32) void k_qkv(
    const __bf16* __restrict__ xb_c, const __bf16* __restrict__ xb_r,
    const __bf16* __restrict__ Wt_c, const __bf16* __restrict__ Wt_r,
    float* __restrict__ qkv_cls, float* __restrict__ qkv_reg) {
  const __bf16* x  = blockIdx.z ? xb_r : xb_c;
  const __bf16* Wt = blockIdx.z ? Wt_r : Wt_c;
  float* out       = blockIdx.z ? qkv_reg : qkv_cls;

  const int nb = blockIdx.x * 16, mb0 = blockIdx.y * 64;
  const int lane = threadIdx.x & 31, half = lane >> 4, l16 = lane & 15;

  v8f acc[4];
#pragma unroll
  for (int t = 0; t < 4; ++t)
    acc[t] = (v8f){0.f, 0.f, 0.f, 0.f, 0.f, 0.f, 0.f, 0.f};

  const __bf16* xrow = x + (size_t)(nb + l16) * CDIM;
#pragma unroll
  for (int kb = 0; kb < CDIM; kb += 32) {
    v16bf a = load_afrag(xrow, kb, half);
#pragma unroll
    for (int t = 0; t < 4; ++t) {
      const int col = mb0 + t * 16 + l16;
      v16bf b = *(const v16bf*)(Wt + (size_t)col * CDIM + kb + 16 * half);
      acc[t] = wmma_bf16(a, b, acc[t]);
    }
  }
#pragma unroll
  for (int t = 0; t < 4; ++t)
#pragma unroll
    for (int r = 0; r < 8; ++r)
      out[(size_t)(nb + r + 8 * half) * TCDIM + mb0 + t * 16 + l16] = acc[t][r];
}

// ---------------------------------------------------------------------------
// Kernel 2: split + L2-normalize. One wave per (mat, head, token), 8 waves/WG.
// ---------------------------------------------------------------------------
__global__ __launch_bounds__(256) void k_norm(
    const float* __restrict__ qkv_c, const float* __restrict__ qkv_r,
    __bf16* __restrict__ qc, __bf16* __restrict__ kc,
    __bf16* __restrict__ qr, __bf16* __restrict__ kr,
    __bf16* __restrict__ mvc, __bf16* __restrict__ mvr,
    __bf16* __restrict__ vTc, __bf16* __restrict__ vTr,
    float* __restrict__ out_xc, float* __restrict__ out_xr) {
  const int w = threadIdx.x >> 5, lane = threadIdx.x & 31;
  const int gid = blockIdx.x * 8 + w;      // 2 * H * N waves
  const int mat = gid >> 14;
  const int h   = (gid >> 12) & 3;
  const int n   = gid & (NTOK - 1);

  const float* qkv = mat ? qkv_r : qkv_c;
  __bf16* qd = mat ? qr : qc;
  __bf16* kd = mat ? kr : kc;
  __bf16* mv = mat ? mvr : mvc;
  __bf16* vT = mat ? vTr : vTc;
  float* outx = mat ? out_xr : out_xc;

  const size_t base = (size_t)n * TCDIM + h * DHEAD + 2 * lane;

  {  // q
    float a0 = qkv[base + 0], a1 = qkv[base + 1];
    float inv = 1.f / fmaxf(sqrtf(rsum32(a0 * a0 + a1 * a1)), 1e-6f);
    __bf16* d = qd + ((size_t)h * NTOK + n) * DHEAD + 2 * lane;
    d[0] = (__bf16)(a0 * inv); d[1] = (__bf16)(a1 * inv);
  }
  {  // k
    float a0 = qkv[base + CDIM + 0], a1 = qkv[base + CDIM + 1];
    float inv = 1.f / fmaxf(sqrtf(rsum32(a0 * a0 + a1 * a1)), 1e-6f);
    __bf16* d = kd + ((size_t)h * NTOK + n) * DHEAD + 2 * lane;
    d[0] = (__bf16)(a0 * inv); d[1] = (__bf16)(a1 * inv);
  }
  {  // v: raw -> out[:,256:512] + raw transposed bf16 + normalized merged bf16
    float a0 = qkv[base + 2 * CDIM + 0], a1 = qkv[base + 2 * CDIM + 1];
    float* o = outx + (size_t)n * (2 * CDIM) + CDIM + h * DHEAD + 2 * lane;
    o[0] = a0; o[1] = a1;
    vT[((size_t)h * DHEAD + 2 * lane + 0) * NTOK + n] = (__bf16)a0;
    vT[((size_t)h * DHEAD + 2 * lane + 1) * NTOK + n] = (__bf16)a1;
    float inv = 1.f / fmaxf(sqrtf(rsum32(a0 * a0 + a1 * a1)), 1e-6f);
    __bf16* d = mv + (size_t)n * CDIM + h * DHEAD + 2 * lane;
    d[0] = (__bf16)(a0 * inv); d[1] = (__bf16)(a1 * inv);
  }
}

// ---------------------------------------------------------------------------
// Kernel 3: fused dual-stream attention, 2 passes with fixed-max softmax.
// grid = N/16 query tiles, block = 256 (8 waves, each owns a 512-col slice).
// ---------------------------------------------------------------------------
__global__ __launch_bounds__(256) void k_attn(
    const __bf16* __restrict__ qc, const __bf16* __restrict__ kc,
    const __bf16* __restrict__ qr, const __bf16* __restrict__ kr,
    const __bf16* __restrict__ vTc, const __bf16* __restrict__ vTr,
    const float* __restrict__ cls_score,
    float* __restrict__ out_xc, float* __restrict__ out_xr,
    float* __restrict__ sim) {
  const int ib = blockIdx.x * 16;
  const int w = threadIdx.x >> 5, lane = threadIdx.x & 31;
  const int half = lane >> 4, l16 = lane & 15;
  const int jw = w * 512;  // this wave's column base

  __shared__ float sL[2][4][8][16];
  __shared__ float gInvL[2][4][16];
  __shared__ __bf16 pstage[8][16][32];
  __shared__ float ored[8][16][64];

  // ---- pass 1: sum_j exp(score - SCALE) per (mat, head, row) --------------
  for (int h = 0; h < HEADS; ++h) {
    for (int mat = 0; mat < 2; ++mat) {
      const __bf16* q = mat ? qr : qc;
      const __bf16* k = mat ? kr : kc;
      const __bf16* qrow = q + ((size_t)h * NTOK + ib + l16) * DHEAD;
      v16bf a0 = load_afrag(qrow, 0, half);
      v16bf a1 = load_afrag(qrow, 32, half);
      float lsum[8];
#pragma unroll
      for (int r = 0; r < 8; ++r) lsum[r] = 0.f;

      for (int jt = 0; jt < 16; ++jt) {
        const int jb = jw + jt * 32;
#pragma unroll
        for (int sub = 0; sub < 2; ++sub) {
          const int jcol = jb + sub * 16 + l16;
          const __bf16* krow = k + ((size_t)h * NTOK + jcol) * DHEAD;
          v8f c = {0.f, 0.f, 0.f, 0.f, 0.f, 0.f, 0.f, 0.f};
          c = wmma_bf16(a0, *(const v16bf*)(krow + 16 * half), c);
          c = wmma_bf16(a1, *(const v16bf*)(krow + 32 + 16 * half), c);
          const float mult = mat ? SCALE_F : SCALE_F * cls_score[jcol];
#pragma unroll
          for (int r = 0; r < 8; ++r)
            lsum[r] += __expf(c[r] * mult - SCALE_F);
        }
      }
#pragma unroll
      for (int r = 0; r < 8; ++r) lsum[r] = rsum16(lsum[r]);
      if (l16 == 0) {
#pragma unroll
        for (int r = 0; r < 8; ++r) sL[mat][h][w][8 * half + r] = lsum[r];
      }
    }
  }
  __syncthreads();
  if (threadIdx.x < 128) {  // 2*4*16 entries
    const int t = threadIdx.x;
    const int mat = t >> 6, hh = (t >> 4) & 3, row = t & 15;
    float L = 0.f;
#pragma unroll
    for (int ww = 0; ww < 8; ++ww) L += sL[mat][hh][ww][row];
    gInvL[mat][hh][row] = 1.f / L;
  }
  __syncthreads();

  // ---- pass 2: probabilities, sim accumulation, P @ V ---------------------
  for (int h = 0; h < HEADS; ++h) {
    const __bf16* qcrow = qc + ((size_t)h * NTOK + ib + l16) * DHEAD;
    const __bf16* qrrow = qr + ((size_t)h * NTOK + ib + l16) * DHEAD;
    v16bf ac0 = load_afrag(qcrow, 0, half), ac1 = load_afrag(qcrow, 32, half);
    v16bf ar0 = load_afrag(qrrow, 0, half), ar1 = load_afrag(qrrow, 32, half);

    float iLc[8], iLr[8];
#pragma unroll
    for (int r = 0; r < 8; ++r) {
      iLc[r] = gInvL[0][h][r + 8 * half];
      iLr[r] = gInvL[1][h][r + 8 * half];
    }
    v8f oc[4], orr[4];
#pragma unroll
    for (int t = 0; t < 4; ++t) {
      oc[t] = (v8f){0.f, 0.f, 0.f, 0.f, 0.f, 0.f, 0.f, 0.f};
      orr[t] = (v8f){0.f, 0.f, 0.f, 0.f, 0.f, 0.f, 0.f, 0.f};
    }

    for (int jt = 0; jt < 16; ++jt) {
      const int jb = jw + jt * 32;
      v8f ccls[2], creg[2];
#pragma unroll
      for (int sub = 0; sub < 2; ++sub) {
        const int jcol = jb + sub * 16 + l16;
        const __bf16* kcrow = kc + ((size_t)h * NTOK + jcol) * DHEAD;
        const __bf16* krrow = kr + ((size_t)h * NTOK + jcol) * DHEAD;
        v8f c0 = {0.f, 0.f, 0.f, 0.f, 0.f, 0.f, 0.f, 0.f};
        c0 = wmma_bf16(ac0, *(const v16bf*)(kcrow + 16 * half), c0);
        c0 = wmma_bf16(ac1, *(const v16bf*)(kcrow + 32 + 16 * half), c0);
        ccls[sub] = c0;
        v8f c1 = {0.f, 0.f, 0.f, 0.f, 0.f, 0.f, 0.f, 0.f};
        c1 = wmma_bf16(ar0, *(const v16bf*)(krrow + 16 * half), c1);
        c1 = wmma_bf16(ar1, *(const v16bf*)(krrow + 32 + 16 * half), c1);
        creg[sub] = c1;
      }
      // probabilities, sim RMW (exclusive element ownership), stage P to LDS
#pragma unroll
      for (int sub = 0; sub < 2; ++sub) {
        const int jcol = jb + sub * 16 + l16;
        const float multc = SCALE_F * cls_score[jcol];
#pragma unroll
        for (int r = 0; r < 8; ++r) {
          float pc = __expf(ccls[sub][r] * multc - SCALE_F) * iLc[r];
          float pr = __expf(creg[sub][r] * SCALE_F - SCALE_F) * iLr[r];
          float p = 0.5f * (pc + pr);
          float* sp = sim + (size_t)(ib + r + 8 * half) * NTOK + jcol;
          if (h == 0) *sp = 0.25f * p;
          else        *sp += 0.25f * p;
          pstage[w][r + 8 * half][sub * 16 + l16] = (__bf16)p;
        }
      }
      // A fragment of P (16x32) from LDS stage (D-layout -> A-layout)
      v16bf ap;
#pragma unroll
      for (int e = 0; e < 16; ++e) {
        int K = e + 16 * half;
        ap[e] = pstage[w][l16][K];
      }
      // O += P @ V  (K = 32 columns of this j-block), both streams
#pragma unroll
      for (int dt = 0; dt < 4; ++dt) {
        const int d = dt * 16 + l16;
        const __bf16* pvc = vTc + ((size_t)h * DHEAD + d) * NTOK + jb + 16 * half;
        const __bf16* pvr = vTr + ((size_t)h * DHEAD + d) * NTOK + jb + 16 * half;
        oc[dt] = wmma_bf16(ap, *(const v16bf*)pvc, oc[dt]);
        orr[dt] = wmma_bf16(ap, *(const v16bf*)pvr, orr[dt]);
      }
    }

    // cross-wave O reduction through LDS (two passes: cls then reg)
#pragma unroll
    for (int pass = 0; pass < 2; ++pass) {
#pragma unroll
      for (int dt = 0; dt < 4; ++dt)
#pragma unroll
        for (int r = 0; r < 8; ++r)
          ored[w][r + 8 * half][dt * 16 + l16] =
              pass ? orr[dt][r] : oc[dt][r];
      __syncthreads();
      float* outp = pass ? out_xr : out_xc;
      for (int idx = threadIdx.x; idx < 16 * 64; idx += 256) {
        const int row = idx >> 6, col = idx & 63;
        float s = 0.f;
#pragma unroll
        for (int ww = 0; ww < 8; ++ww) s += ored[ww][row][col];
        outp[(size_t)(ib + row) * (2 * CDIM) + h * DHEAD + col] = s;
      }
      __syncthreads();
    }
  }
}

// ---------------------------------------------------------------------------
// Kernel 4: row sum of exp(sim - 1) (sim values are in (0,1]).
// ---------------------------------------------------------------------------
__global__ __launch_bounds__(256) void k_simstat(
    const float* __restrict__ sim, float* __restrict__ rowInvL) {
  const int row = blockIdx.x;
  const int w = threadIdx.x >> 5, lane = threadIdx.x & 31;
  const float* p = sim + (size_t)row * NTOK;
  float s = 0.f;
#pragma unroll
  for (int i = 0; i < 16; ++i) s += __expf(p[threadIdx.x + i * 256] - 1.0f);
  s = rsum32(s);
  __shared__ float sl[8];
  if (lane == 0) sl[w] = s;
  __syncthreads();
  if (threadIdx.x == 0) {
    float L = 0.f;
#pragma unroll
    for (int i = 0; i < 8; ++i) L += sl[i];
    rowInvL[row] = 1.f / L;
  }
}

// ---------------------------------------------------------------------------
// Kernel 5: Gram matrices (v_n · v_n^T)/H via WMMA (K=256), thresholds into
// mask bytes, softmax(sim) in place, per-row denom = sum(mask * soft).
// grid = N/16 row tiles, block = 256 (8 waves x 512-col slices).
// ---------------------------------------------------------------------------
__global__ __launch_bounds__(256) void k_gram(
    const __bf16* __restrict__ mvc, const __bf16* __restrict__ mvr,
    const float* __restrict__ rowInvL,
    float* __restrict__ sim, unsigned char* __restrict__ maskb,
    float* __restrict__ denom) {
  const int ib = blockIdx.x * 16;
  const int w = threadIdx.x >> 5, lane = threadIdx.x & 31;
  const int half = lane >> 4, l16 = lane & 15;

  v16bf Ac[8], Ar[8];
  {
    const __bf16* rc = mvc + (size_t)(ib + l16) * CDIM;
    const __bf16* rr = mvr + (size_t)(ib + l16) * CDIM;
#pragma unroll
    for (int kc8 = 0; kc8 < 8; ++kc8) {
      Ac[kc8] = load_afrag(rc, kc8 * 32, half);
      Ar[kc8] = load_afrag(rr, kc8 * 32, half);
    }
  }
  float iL[8], dacc[8];
#pragma unroll
  for (int r = 0; r < 8; ++r) {
    iL[r] = rowInvL[ib + r + 8 * half];
    dacc[r] = 0.f;
  }

  for (int jt = 0; jt < 32; ++jt) {
    const int jcol = w * 512 + jt * 16 + l16;
    const __bf16* bc = mvc + (size_t)jcol * CDIM;
    const __bf16* br = mvr + (size_t)jcol * CDIM;
    v8f cc = {0.f, 0.f, 0.f, 0.f, 0.f, 0.f, 0.f, 0.f};
    v8f cr = {0.f, 0.f, 0.f, 0.f, 0.f, 0.f, 0.f, 0.f};
#pragma unroll
    for (int kc8 = 0; kc8 < 8; ++kc8) {
      cc = wmma_bf16(Ac[kc8], *(const v16bf*)(bc + kc8 * 32 + 16 * half), cc);
      cr = wmma_bf16(Ar[kc8], *(const v16bf*)(br + kc8 * 32 + 16 * half), cr);
    }
#pragma unroll
    for (int r = 0; r < 8; ++r) {
      const float gc = cc[r] * 0.25f;
      const float gr = cr[r] * 0.25f;
      const size_t off = (size_t)(ib + r + 8 * half) * NTOK + jcol;
      const float soft = __expf(sim[off] - 1.0f) * iL[r];
      sim[off] = soft;
      unsigned char mb =
          (unsigned char)((gc > 0.75f ? 1 : 0) | (gr > 0.99f ? 2 : 0));
      maskb[off] = mb;
      if (mb & 1) dacc[r] += soft;
    }
  }
  __shared__ float sden[8][16];
#pragma unroll
  for (int r = 0; r < 8; ++r) dacc[r] = rsum16(dacc[r]);
  if (l16 == 0) {
#pragma unroll
    for (int r = 0; r < 8; ++r) sden[w][8 * half + r] = dacc[r];
  }
  __syncthreads();
  if (threadIdx.x < 16) {
    float s = 0.f;
#pragma unroll
    for (int ww = 0; ww < 8; ++ww) s += sden[ww][threadIdx.x];
    denom[ib + threadIdx.x] = s;
  }
}

// ---------------------------------------------------------------------------
// Kernel 6: finalize. sim_round2 = mask*soft/denom.
// denom2 = sum(mask*sim_round2) == denom/denom == 1 -> obj = obj_bit*sim_round2
// ---------------------------------------------------------------------------
__global__ __launch_bounds__(256) void k_final(
    float* __restrict__ sim, float* __restrict__ obj,
    const unsigned char* __restrict__ maskb, const float* __restrict__ denom) {
  const int row = blockIdx.x;
  const float inv = 1.f / denom[row];
#pragma unroll
  for (int i = 0; i < 16; ++i) {
    const int col = threadIdx.x + i * 256;
    const size_t off = (size_t)row * NTOK + col;
    const float soft = sim[off];
    const unsigned char mb = maskb[off];
    const float sf = (mb & 1) ? soft * inv : 0.f;
    sim[off] = sf;
    obj[off] = (mb & 2) ? sf : 0.f;
  }
}

// ---------------------------------------------------------------------------
extern "C" void kernel_launch(void* const* d_in, const int* in_sizes, int n_in,
                              void* d_out, int out_size, void* d_ws,
                              size_t ws_size, hipStream_t stream) {
  (void)in_sizes; (void)n_in; (void)out_size; (void)ws_size;
  const float* x_cls = (const float*)d_in[0];
  const float* x_reg = (const float*)d_in[1];
  const float* cls_score = (const float*)d_in[2];
  const float* W_cls = (const float*)d_in[3];
  const float* W_reg = (const float*)d_in[4];

  float* out = (float*)d_out;
  float* out_xc = out;
  float* out_xr = out + (size_t)NTOK * 2 * CDIM;
  float* sim = out + (size_t)2 * NTOK * 2 * CDIM;
  float* obj = sim + (size_t)NTOK * NTOK;

  // workspace layout (256B aligned carve-out)
  char* ws = (char*)d_ws;
  size_t off = 0;
  auto alloc = [&](size_t bytes) {
    char* p = ws + off;
    off += (bytes + 255) & ~(size_t)255;
    return p;
  };
  float*  qkv_c = (float*)alloc((size_t)NTOK * TCDIM * 4);
  float*  qkv_r = (float*)alloc((size_t)NTOK * TCDIM * 4);
  __bf16* xb_c  = (__bf16*)alloc((size_t)NTOK * CDIM * 2);
  __bf16* xb_r  = (__bf16*)alloc((size_t)NTOK * CDIM * 2);
  __bf16* Wt_c  = (__bf16*)alloc((size_t)TCDIM * CDIM * 2);
  __bf16* Wt_r  = (__bf16*)alloc((size_t)TCDIM * CDIM * 2);
  __bf16* qc = (__bf16*)alloc((size_t)HEADS * NTOK * DHEAD * 2);
  __bf16* kc = (__bf16*)alloc((size_t)HEADS * NTOK * DHEAD * 2);
  __bf16* qr = (__bf16*)alloc((size_t)HEADS * NTOK * DHEAD * 2);
  __bf16* kr = (__bf16*)alloc((size_t)HEADS * NTOK * DHEAD * 2);
  __bf16* mvc = (__bf16*)alloc((size_t)NTOK * CDIM * 2);
  __bf16* mvr = (__bf16*)alloc((size_t)NTOK * CDIM * 2);
  __bf16* vTc = (__bf16*)alloc((size_t)HEADS * DHEAD * NTOK * 2);
  __bf16* vTr = (__bf16*)alloc((size_t)HEADS * DHEAD * NTOK * 2);
  float* rowInvL = (float*)alloc((size_t)NTOK * 4);
  float* denom   = (float*)alloc((size_t)NTOK * 4);
  unsigned char* maskb = (unsigned char*)alloc((size_t)NTOK * NTOK);

  k_cvt_x<<<2 * NTOK * CDIM / 256, 256, 0, stream>>>(x_cls, x_reg, xb_c, xb_r);
  k_cvt_wt<<<2 * TCDIM * CDIM / 256, 256, 0, stream>>>(W_cls, W_reg, Wt_c, Wt_r);
  k_qkv<<<dim3(NTOK / 16, TCDIM / 64, 2), 32, 0, stream>>>(
      xb_c, xb_r, Wt_c, Wt_r, qkv_c, qkv_r);
  k_norm<<<2 * HEADS * NTOK / 8, 256, 0, stream>>>(
      qkv_c, qkv_r, qc, kc, qr, kr, mvc, mvr, vTc, vTr, out_xc, out_xr);
  k_attn<<<NTOK / 16, 256, 0, stream>>>(
      qc, kc, qr, kr, vTc, vTr, cls_score, out_xc, out_xr, sim);
  k_simstat<<<NTOK, 256, 0, stream>>>(sim, rowInvL);
  k_gram<<<NTOK / 16, 256, 0, stream>>>(mvc, mvr, rowInvL, sim, maskb, denom);
  k_final<<<NTOK, 256, 0, stream>>>(sim, obj, maskb, denom);
}
